// BasicGAT_3942779978210
// MI455X (gfx1250) — matrix operand order repaired and checked by reading, not compiled, not run on hardware
//
#include <hip/hip_runtime.h>

// ---------------- problem constants ----------------
#define BB    4
#define NN    4096
#define FF    256
#define HH    256
#define EE    262144
#define LL    3
#define NTOT  (BB * NN)            // 16384
#define ETOT  (EE + NTOT)          // 278528 (with self loops)
#define NEG_SLOPE 0.2f
#define EPS 1e-5f

#define MT    (NTOT / 16)          // 1024 M tiles
#define KTC   24                   // conv K tiles (K = 768)
#define KTG   8                    // gat  K tiles (K = 256)
#define NTI   (HH / 16)            // 16 N tiles

typedef __attribute__((ext_vector_type(16))) __bf16 v16bf;
typedef __attribute__((ext_vector_type(8)))  float  v8f;

// ---------------- helpers ----------------
__device__ __forceinline__ void atomicMaxFloat(float* addr, float val) {
    if (val >= 0.0f) {
        atomicMax((int*)addr, __float_as_int(val));
    } else {
        atomicMin((unsigned int*)addr, (unsigned int)__float_as_int(val));
    }
}

__device__ __forceinline__ unsigned short f2bf_bits(float f) {
    __bf16 b = (__bf16)f;
    unsigned short s;
    __builtin_memcpy(&s, &b, 2);
    return s;
}

// A-fragment K base for VGPR v, lane half (16-bit A 16x32 ISA layout)
__device__ __forceinline__ int afrag_kbase(int v, int half) {
    return (v < 4 ? 2 * v : 16 + 2 * (v - 4)) + (half ? 8 : 0);
}

// Inverse map: (local row m, local K kl) -> (lane, vgpr) within one A fragment
__device__ __forceinline__ void afrag_inv(int m, int kl, int& lane, int& v) {
    if (kl < 8)       { lane = m;      v = kl >> 1; }
    else if (kl < 16) { lane = m + 16; v = (kl - 8) >> 1; }
    else if (kl < 24) { lane = m;      v = 4 + ((kl - 16) >> 1); }
    else              { lane = m + 16; v = 4 + ((kl - 24) >> 1); }
}

// ---------------- pack im2col A for conv into A-fragment layout ----------------
// Apack layout: [tm=1024][kt=24][lane=32][v=8] u32; element K of row maps to
// tap d = K/256, feature f = K%256, source node row+d-1 (zero-padded).
__global__ void pack_aconv_kernel(const float* __restrict__ x, unsigned* __restrict__ apack) {
    int t = blockIdx.x * blockDim.x + threadIdx.x;  // MT*KTC*32*8 = 6291456
    if (t >= MT * KTC * 32 * 8) return;
    int v    = t & 7;
    int lane = (t >> 3) & 31;
    int kt   = (t >> 8) % KTC;
    int tm   = t / (KTC * 256);
    int m    = lane & 15;
    int row  = tm * 16 + m;
    int K0   = kt * 32 + afrag_kbase(v, lane >> 4);  // even; K0,K0+1 share tap d
    int d    = K0 >> 8;
    int f    = K0 & 255;
    int npos = (row & (NN - 1)) + d - 1;
    float w0 = 0.0f, w1 = 0.0f;
    if (npos >= 0 && npos < NN) {
        const float* src = x + (size_t)(row + d - 1) * FF;
        w0 = src[f];
        w1 = src[f + 1];
    }
    apack[t] = ((unsigned)f2bf_bits(w1) << 16) | (unsigned)f2bf_bits(w0);
}

// ---------------- pack conv weights into WMMA B-fragment layout ----------------
// Bpack layout: [kt=24][ntile=16][lane=32][v=8] u32; B(K,n) = tc_w[h=n][f=K%256][d=K/256]
__global__ void pack_bconv_kernel(const float* __restrict__ tcw, unsigned* __restrict__ bpack) {
    int t = blockIdx.x * blockDim.x + threadIdx.x;  // KTC*NTI*32*8 = 98304
    if (t >= KTC * NTI * 32 * 8) return;
    int v     = t & 7;
    int lane  = (t >> 3) & 31;
    int ntile = (t >> 8) % NTI;
    int kt    = t / (NTI * 256);
    int n = ntile * 16 + (lane & 15);
    int K = kt * 32 + afrag_kbase(v, lane >> 4);  // even; K,K+1 share tap d
    int d = K >> 8;
    int f = K & 255;
    float w0 = tcw[n * (FF * 3) + f * 3 + d];
    float w1 = tcw[n * (FF * 3) + (f + 1) * 3 + d];
    bpack[t] = ((unsigned)f2bf_bits(w1) << 16) | (unsigned)f2bf_bits(w0);
}

// gat_W[l] is [K=H_in][N=H_out] row-major -> Bpack [l][kt=8][ntile=16][lane=32][v=8]
__global__ void pack_bgat_kernel(const float* __restrict__ gw, unsigned* __restrict__ bpack) {
    int t = blockIdx.x * blockDim.x + threadIdx.x;  // LL*KTG*NTI*32*8 = 98304
    if (t >= LL * KTG * NTI * 32 * 8) return;
    int v     = t & 7;
    int lane  = (t >> 3) & 31;
    int ntile = (t >> 8) & 15;
    int kt    = (t >> 12) & 7;
    int l     = t >> 15;
    int n = ntile * 16 + (lane & 15);
    int K = kt * 32 + afrag_kbase(v, lane >> 4);
    const float* W = gw + (size_t)l * HH * HH;
    float w0 = W[K * HH + n];
    float w1 = W[(K + 1) * HH + n];
    bpack[t] = ((unsigned)f2bf_bits(w1) << 16) | (unsigned)f2bf_bits(w0);
}

// ---------------- conv GEMM: C[16384x256] = Apack(im2col, K=768) x Bpack ----------------
// one wave computes a 64x64 tile (4 M-subtiles x 4 N-subtiles = 16 accumulators);
// per K-step: 8 A loads + 8 B loads feed 16 WMMAs. Epilogue: +bias +residual, relu,
// writes result as packed bf16 A-fragments for the first GAT GEMM.
__global__ __launch_bounds__(32)
void conv_gemm_kernel(const unsigned* __restrict__ apack, const unsigned* __restrict__ bpack,
                      const float* __restrict__ tcb, const float* __restrict__ xres,
                      unsigned short* __restrict__ apack_out) {
    const int lane = threadIdx.x;
    const int tm0 = blockIdx.x * 4;   // M subtile base (0..1020)
    const int tn0 = blockIdx.y * 4;   // N subtile base (0..12)
    const int half = lane >> 4;
    const int mrow = lane & 15;

    v8f c[4][4];
#pragma unroll
    for (int mi = 0; mi < 4; ++mi)
#pragma unroll
        for (int nj = 0; nj < 4; ++nj)
#pragma unroll
            for (int i = 0; i < 8; ++i) c[mi][nj][i] = 0.0f;

    for (int kt = 0; kt < KTC; ++kt) {
        union { v16bf v; uint4 q[2]; } a[4];
#pragma unroll
        for (int mi = 0; mi < 4; ++mi) {
            const uint4* ap = (const uint4*)(apack + (((size_t)(tm0 + mi) * KTC + kt) * 32 + lane) * 8);
            a[mi].q[0] = ap[0];
            a[mi].q[1] = ap[1];
        }
#pragma unroll
        for (int nj = 0; nj < 4; ++nj) {
            union { v16bf v; uint4 q[2]; } b;
            const uint4* bp = (const uint4*)(bpack + (((size_t)kt * NTI + (tn0 + nj)) * 32 + lane) * 8);
            b.q[0] = bp[0];
            b.q[1] = bp[1];
#pragma unroll
            for (int mi = 0; mi < 4; ++mi) {
                c[mi][nj] = __builtin_amdgcn_wmma_f32_16x16x32_bf16(false, a[mi].v, false, b.v,
                                                                    (short)0, c[mi][nj], false, false);
            }
        }
    }
#pragma unroll
    for (int mi = 0; mi < 4; ++mi) {
#pragma unroll
        for (int nj = 0; nj < 4; ++nj) {
#pragma unroll
            for (int v = 0; v < 8; ++v) {
                int rloc = v + half * 8;
                int row = (tm0 + mi) * 16 + rloc;
                int col = (tn0 + nj) * 16 + mrow;
                float val = c[mi][nj][v] + tcb[col] + xres[(size_t)row * HH + col];
                val = val > 0.0f ? val : 0.0f;
                int kt2 = col >> 5, kl = col & 31, lane2, v2;
                afrag_inv(rloc, kl, lane2, v2);
                size_t si = ((((size_t)(tm0 + mi) * KTG + kt2) * 32 + lane2) * 8 + v2) * 2 + (kl & 1);
                apack_out[si] = f2bf_bits(val);
            }
        }
    }
}

// ---------------- GAT linear: hw = h @ W (packed A, packed B, f32 out), 64x64 tile ----
__global__ __launch_bounds__(32)
void gat_gemm_kernel(const unsigned* __restrict__ apack, const unsigned* __restrict__ bpack,
                     float* __restrict__ Cout) {
    const int lane = threadIdx.x;
    const int tm0 = blockIdx.x * 4;
    const int tn0 = blockIdx.y * 4;
    const int half = lane >> 4;
    const int mrow = lane & 15;

    v8f c[4][4];
#pragma unroll
    for (int mi = 0; mi < 4; ++mi)
#pragma unroll
        for (int nj = 0; nj < 4; ++nj)
#pragma unroll
            for (int i = 0; i < 8; ++i) c[mi][nj][i] = 0.0f;

    for (int kt = 0; kt < KTG; ++kt) {
        union { v16bf v; uint4 q[2]; } a[4];
#pragma unroll
        for (int mi = 0; mi < 4; ++mi) {
            const uint4* ap = (const uint4*)(apack + (((size_t)(tm0 + mi) * KTG + kt) * 32 + lane) * 8);
            a[mi].q[0] = ap[0];
            a[mi].q[1] = ap[1];
        }
#pragma unroll
        for (int nj = 0; nj < 4; ++nj) {
            union { v16bf v; uint4 q[2]; } b;
            const uint4* bp = (const uint4*)(bpack + (((size_t)kt * NTI + (tn0 + nj)) * 32 + lane) * 8);
            b.q[0] = bp[0];
            b.q[1] = bp[1];
#pragma unroll
            for (int mi = 0; mi < 4; ++mi) {
                c[mi][nj] = __builtin_amdgcn_wmma_f32_16x16x32_bf16(false, a[mi].v, false, b.v,
                                                                    (short)0, c[mi][nj], false, false);
            }
        }
    }
#pragma unroll
    for (int mi = 0; mi < 4; ++mi) {
#pragma unroll
        for (int nj = 0; nj < 4; ++nj) {
#pragma unroll
            for (int v = 0; v < 8; ++v) {
                int row = (tm0 + mi) * 16 + v + half * 8;
                int col = (tn0 + nj) * 16 + mrow;
                Cout[(size_t)row * HH + col] = c[mi][nj][v];
            }
        }
    }
}

// ---------------- per-node attention dots: es = hw . a_s, ed = hw . a_d ----------------
__global__ void dots_kernel(const float* __restrict__ hw, const float* __restrict__ as,
                            const float* __restrict__ ad, float* __restrict__ es,
                            float* __restrict__ ed) {
    int node = blockIdx.x * 8 + (threadIdx.x >> 5);
    int t = threadIdx.x & 31;
    float ss = 0.0f, sd = 0.0f;
#pragma unroll
    for (int j = 0; j < 8; ++j) {
        float v = hw[(size_t)node * HH + t + 32 * j];
        ss += v * as[t + 32 * j];
        sd += v * ad[t + 32 * j];
    }
#pragma unroll
    for (int off = 16; off > 0; off >>= 1) {
        ss += __shfl_xor(ss, off, 32);
        sd += __shfl_xor(sd, off, 32);
    }
    if (t == 0) { es[node] = ss; ed[node] = sd; }
}

// ---------------- softmax scratch init ----------------
__global__ void init_softmax_kernel(float* __restrict__ mmax, float* __restrict__ den) {
    int i = blockIdx.x * blockDim.x + threadIdx.x;
    if (i < NTOT) { mmax[i] = -__builtin_inff(); den[i] = 0.0f; }
}

// ---------------- edge logits + segment max ----------------
__global__ void edge_logits_kernel(const long long* __restrict__ ei, const float* __restrict__ es,
                                   const float* __restrict__ ed, float* __restrict__ ev,
                                   float* __restrict__ mmax) {
    int e = blockIdx.x * blockDim.x + threadIdx.x;
    if (e >= ETOT) return;
    int s, d;
    if (e < EE) { s = (int)ei[e]; d = (int)ei[EE + e]; } else { s = d = e - EE; }
    float v = es[s] + ed[d];
    v = v > 0.0f ? v : NEG_SLOPE * v;
    ev[e] = v;
    atomicMaxFloat(&mmax[d], v);
}

// ---------------- edge exp + segment sum ----------------
__global__ void edge_exp_kernel(const long long* __restrict__ ei, const float* __restrict__ ev,
                                const float* __restrict__ mmax, float* __restrict__ eex,
                                float* __restrict__ den) {
    int e = blockIdx.x * blockDim.x + threadIdx.x;
    if (e >= ETOT) return;
    int d;
    if (e < EE) { d = (int)ei[EE + e]; } else { d = e - EE; }
    float ex = __expf(ev[e] - mmax[d]);
    eex[e] = ex;
    atomicAdd(&den[d], ex);
}

// ---------------- aggregation: agg[dst,:] += alpha * hw[src,:] ----------------
__global__ void edge_agg_kernel(const long long* __restrict__ ei, const float* __restrict__ eex,
                                const float* __restrict__ den, const float* __restrict__ hw,
                                float* __restrict__ agg) {
    int tid = blockIdx.x * blockDim.x + threadIdx.x;   // ETOT*64 threads, 4 floats each
    int e = tid >> 6;
    if (e >= ETOT) return;
    int c = (tid & 63) << 2;
    int s, d;
    if (e < EE) { s = (int)ei[e]; d = (int)ei[EE + e]; } else { s = d = e - EE; }
    float alpha = eex[e] / den[d];
    float4 hv = *(const float4*)(hw + (size_t)s * HH + c);
    float* ap = agg + (size_t)d * HH + c;
    atomicAdd(ap + 0, alpha * hv.x);
    atomicAdd(ap + 1, alpha * hv.y);
    atomicAdd(ap + 2, alpha * hv.z);
    atomicAdd(ap + 3, alpha * hv.w);
}

// ---------------- layer epilogue: h = relu(agg + b); writes f32 + packed bf16 A-frag --
__global__ void layer_epilogue_kernel(const float* __restrict__ agg, const float* __restrict__ bias,
                                      float* __restrict__ h, unsigned short* __restrict__ apack_out) {
    int i = blockIdx.x * blockDim.x + threadIdx.x;
    if (i >= NTOT * HH) return;
    int row = i >> 8;
    int col = i & 255;
    float v = agg[i] + bias[col];
    v = v > 0.0f ? v : 0.0f;
    h[i] = v;
    int tm = row >> 4, m = row & 15;
    int kt = col >> 5, kl = col & 31, lane, vv;
    afrag_inv(m, kl, lane, vv);
    size_t si = ((((size_t)tm * KTG + kt) * 32 + lane) * 8 + vv) * 2 + (kl & 1);
    apack_out[si] = f2bf_bits(v);
}

// ---------------- final conv at position 0 only (taps k=1,2 reach n=0,1) ----------------
__global__ void final_conv_kernel(const float* __restrict__ h, const float* __restrict__ tcw,
                                  const float* __restrict__ tcb, float* __restrict__ outp) {
    int t = blockIdx.x * blockDim.x + threadIdx.x;   // B*H = 1024
    if (t >= BB * HH) return;
    int b = t >> 8;
    int hh = t & 255;
    float acc = tcb[hh];
    const float* r0 = h + (size_t)(b * NN + 0) * HH;
    const float* r1 = h + (size_t)(b * NN + 1) * HH;
    for (int f = 0; f < HH; ++f) {
        acc += tcw[hh * (FF * 3) + f * 3 + 1] * r0[f];
        acc += tcw[hh * (FF * 3) + f * 3 + 2] * r1[f];
    }
    outp[t] = acc;
}

// ---------------- layernorm over H per batch + relu ----------------
__global__ void ln_kernel(const float* __restrict__ outp, const float* __restrict__ g,
                          const float* __restrict__ bb, float* __restrict__ out) {
    __shared__ float red[HH];
    int b = blockIdx.x;
    int t = threadIdx.x;
    float v = outp[b * HH + t];
    red[t] = v;
    __syncthreads();
    for (int s = HH / 2; s > 0; s >>= 1) {
        if (t < s) red[t] += red[t + s];
        __syncthreads();
    }
    float mu = red[0] / (float)HH;
    __syncthreads();
    float dv = v - mu;
    red[t] = dv * dv;
    __syncthreads();
    for (int s = HH / 2; s > 0; s >>= 1) {
        if (t < s) red[t] += red[t + s];
        __syncthreads();
    }
    float var = red[0] / (float)HH;
    float o = dv * rsqrtf(var + EPS) * g[t] + bb[t];
    out[b * HH + t] = o > 0.0f ? o : 0.0f;
}

// ---------------- host launch ----------------
extern "C" void kernel_launch(void* const* d_in, const int* in_sizes, int n_in,
                              void* d_out, int out_size, void* d_ws, size_t ws_size,
                              hipStream_t stream) {
    const float*     x     = (const float*)d_in[0];
    const long long* ei    = (const long long*)d_in[1];
    const float*     tcw   = (const float*)d_in[2];
    const float*     tcb   = (const float*)d_in[3];
    const float*     gatW  = (const float*)d_in[4];
    const float*     gatas = (const float*)d_in[5];
    const float*     gatad = (const float*)d_in[6];
    const float*     gatb  = (const float*)d_in[7];
    const float*     lng   = (const float*)d_in[8];
    const float*     lnb   = (const float*)d_in[9];
    float* out = (float*)d_out;

    char* ws = (char*)d_ws;
    size_t off = 0;
    auto carve = [&](size_t bytes) { char* p = ws + off; off += (bytes + 255) & ~(size_t)255; return p; };

    unsigned* apackC = (unsigned*)carve((size_t)MT * KTC * 32 * 8 * 4);   // 25.2 MB
    unsigned* apackG = (unsigned*)carve((size_t)MT * KTG * 32 * 8 * 4);   // 8.4 MB
    unsigned* bpc    = (unsigned*)carve((size_t)KTC * NTI * 32 * 8 * 4);  // 393 KB
    unsigned* bpg    = (unsigned*)carve((size_t)LL * KTG * NTI * 32 * 8 * 4);
    float*    h      = (float*)  carve((size_t)NTOT * HH * 4);
    float*    hw     = (float*)  carve((size_t)NTOT * HH * 4);
    float*    agg    = (float*)  carve((size_t)NTOT * HH * 4);
    float*    es     = (float*)  carve((size_t)NTOT * 4);
    float*    ed     = (float*)  carve((size_t)NTOT * 4);
    float*    mmax   = (float*)  carve((size_t)NTOT * 4);
    float*    den    = (float*)  carve((size_t)NTOT * 4);
    float*    ev     = (float*)  carve((size_t)ETOT * 4);
    float*    eex    = (float*)  carve((size_t)ETOT * 4);
    float*    outp   = (float*)  carve((size_t)BB * HH * 4);

    // 1) pack A (im2col, 3 taps, zero-padded) and all weight B matrices
    pack_aconv_kernel<<<(MT * KTC * 32 * 8 + 255) / 256, 256, 0, stream>>>(x, apackC);
    pack_bconv_kernel<<<(KTC * NTI * 32 * 8 + 255) / 256, 256, 0, stream>>>(tcw, bpc);
    pack_bgat_kernel<<<(LL * KTG * NTI * 32 * 8 + 255) / 256, 256, 0, stream>>>(gatW, bpg);

    // 2) fused temporal conv GEMM (K=768) + residual + relu -> packed A for GAT layer 0
    conv_gemm_kernel<<<dim3(MT / 4, NTI / 4), 32, 0, stream>>>(apackC, bpc, tcb, x,
                                                               (unsigned short*)apackG);

    // 3) GAT layers
    for (int l = 0; l < LL; ++l) {
        gat_gemm_kernel<<<dim3(MT / 4, NTI / 4), 32, 0, stream>>>(
            apackG, bpg + (size_t)l * KTG * NTI * 32 * 8, hw);
        dots_kernel<<<NTOT / 8, 256, 0, stream>>>(hw, gatas + l * HH, gatad + l * HH, es, ed);
        init_softmax_kernel<<<NTOT / 256, 256, 0, stream>>>(mmax, den);
        edge_logits_kernel<<<(ETOT + 255) / 256, 256, 0, stream>>>(ei, es, ed, ev, mmax);
        edge_exp_kernel<<<(ETOT + 255) / 256, 256, 0, stream>>>(ei, ev, mmax, eex, den);
        hipMemsetAsync(agg, 0, (size_t)NTOT * HH * 4, stream);
        edge_agg_kernel<<<(ETOT * 64) / 256, 256, 0, stream>>>(ei, eex, den, hw, agg);
        layer_epilogue_kernel<<<(NTOT * HH + 255) / 256, 256, 0, stream>>>(
            agg, gatb + l * HH, h, (unsigned short*)apackG);
    }

    // 4) final conv (position 0 only) + layernorm + relu
    final_conv_kernel<<<(BB * HH + 255) / 256, 256, 0, stream>>>(h, tcw, tcb, outp);
    ln_kernel<<<BB, HH, 0, stream>>>(outp, lng, lnb, out);
}
